// SentenceDecoder_3470333575171
// MI455X (gfx1250) — compile-verified, matrix operand order checked by reading, not run on hardware
//
#include <hip/hip_runtime.h>

#define MAX_SENT 120
#define MAX_WORD 10
#define EMB 80
#define EMB_P 96
#define WDEC 300
#define WDEC_P 320
#define SDEC 2800
#define SDEC_P 2816
#define VOCAB 4000
#define BATCH 32

typedef __attribute__((ext_vector_type(16))) __bf16 v16bf;
typedef __attribute__((ext_vector_type(8)))  float  v8f;
typedef unsigned short u16;
typedef unsigned int   u32;
typedef unsigned long long u64;

__device__ __forceinline__ u16 f2bf(float f) {
  u32 u = __float_as_uint(f);
  u32 r = u + 0x7FFFu + ((u >> 16) & 1u);   // round-to-nearest-even
  return (u16)(r >> 16);
}
__device__ __forceinline__ float bf2f(u16 h) {
  return __uint_as_float(((u32)h) << 16);
}

// Load a 16x32 bf16 fragment in the CDNA5 WMMA 16-bit layout.
// A-matrix: row = row0 + (lane&15). B-matrix from row-major W(N,K) has the
// same per-lane layout with row = n0 + (lane&15) (B[k][n] = W[n][k]).
// Half-pair j (one dword) holds K = k, k+1 with
//   k = (2j<8 ? 2j : 2j+8) + (lane>=16 ? 8 : 0)
__device__ __forceinline__ v16bf load_frag(const u16* base, int ldk, int row0,
                                           int k0, int lane) {
  union { v16bf v; u32 u[8]; } r;
  const u16* p = base + (size_t)(row0 + (lane & 15)) * ldk + k0;
#pragma unroll
  for (int j = 0; j < 8; ++j) {
    int i2 = 2 * j;
    int k = (i2 < 8 ? i2 : i2 + 8) + ((lane & 16) ? 8 : 0);
    r.u[j] = *(const u32*)(p + k);
  }
  return r.v;
}

#define WMMA_BF16(A, Bm, C) \
  __builtin_amdgcn_wmma_f32_16x16x32_bf16(false, (A), false, (Bm), (short)0, (C), false, false)

// ---------------- one-time weight conversion (f32 -> padded bf16) ----------
__global__ void convert_pad(const float* __restrict__ src, int rows, int cols,
                            u16* __restrict__ dst, int rows_p, int cols_p) {
  size_t total = (size_t)rows_p * cols_p;
  for (size_t i = (size_t)blockIdx.x * blockDim.x + threadIdx.x; i < total;
       i += (size_t)gridDim.x * blockDim.x) {
    int r = (int)(i / cols_p), c = (int)(i % cols_p);
    float v = (r < rows && c < cols) ? src[(size_t)r * cols + c] : 0.f;
    dst[i] = f2bf(v);
  }
}

// gate-major source (3*rpg, cols) -> (3, rpg_p, cols_p) zero padded per gate
__global__ void convert_pad_gates(const float* __restrict__ src, int rpg, int cols,
                                  u16* __restrict__ dst, int rpg_p, int cols_p) {
  size_t total = (size_t)3 * rpg_p * cols_p;
  for (size_t i = (size_t)blockIdx.x * blockDim.x + threadIdx.x; i < total;
       i += (size_t)gridDim.x * blockDim.x) {
    int cp = (int)(i % cols_p);
    size_t t = i / cols_p;
    int rp = (int)(t % rpg_p);
    int g  = (int)(t / rpg_p);
    float v = (rp < rpg && cp < cols) ? src[((size_t)g * rpg + rp) * cols + cp] : 0.f;
    dst[i] = f2bf(v);
  }
}

__global__ void init_state(const float* __restrict__ hidden, float* __restrict__ hf,
                           u16* __restrict__ hbf, u16* __restrict__ xs) {
  for (int i = blockIdx.x * blockDim.x + threadIdx.x; i < BATCH * SDEC_P;
       i += gridDim.x * blockDim.x) {
    int b = i / SDEC_P, n = i % SDEC_P;
    float v = (n < SDEC) ? hidden[(size_t)b * SDEC + n] : 0.f;
    hf[i] = v;
    hbf[i] = f2bf(v);
  }
  for (int i = blockIdx.x * blockDim.x + threadIdx.x; i < BATCH * WDEC_P;
       i += gridDim.x * blockDim.x) {
    int n = i % WDEC_P;
    xs[i] = (n < WDEC) ? (u16)0x3F80 : (u16)0;  // 1.0 bf16
  }
}

// ---------------- sentence GRU step (fused gates, N-tile per block) --------
// block = 256 threads (8 waves); waves 0..5 = {ih,hh} x {r,z,n}; 2 M-tiles each
__global__ void sent_gru_step(const u16* __restrict__ x_bf,   // 32 x 320
                              const u16* __restrict__ h_bf,   // 32 x 2816
                              const float* __restrict__ h_f32,// 32 x 2816
                              const u16* __restrict__ Wih,    // 8400 x 320
                              const u16* __restrict__ Whh,    // 8400 x 2816
                              const float* __restrict__ bih,  // 8400
                              const float* __restrict__ bhh,  // 8400
                              float* __restrict__ h_f32_new,
                              u16* __restrict__ h_bf_new) {
  __shared__ float accs[6][BATCH][16];
  int n0 = blockIdx.x * 16;                 // 0..2784
  int wave = threadIdx.x >> 5, lane = threadIdx.x & 31;
  if (wave < 6) {
    int src = wave / 3, gate = wave % 3;
    const u16* W; const u16* X; int ldk; int nk;
    if (src == 0) { W = Wih + (size_t)(gate * SDEC + n0) * WDEC_P; X = x_bf; ldk = WDEC_P; nk = WDEC_P / 32; }
    else          { W = Whh + (size_t)(gate * SDEC + n0) * SDEC_P; X = h_bf; ldk = SDEC_P; nk = SDEC_P / 32; }
    v8f a0 = {}, a1 = {};
    for (int kc = 0; kc < nk; ++kc) {
      int k0 = kc * 32;
      v16bf fb = load_frag(W, ldk, 0, k0, lane);
      v16bf fa = load_frag(X, ldk, 0, k0, lane);
      a0 = WMMA_BF16(fa, fb, a0);
      fa = load_frag(X, ldk, 16, k0, lane);
      a1 = WMMA_BF16(fa, fb, a1);
    }
    int col = lane & 15, rb = (lane & 16) ? 8 : 0;
#pragma unroll
    for (int j = 0; j < 8; ++j) {
      accs[wave][rb + j][col]      = a0[j];
      accs[wave][16 + rb + j][col] = a1[j];
    }
  }
  __syncthreads();
  for (int e = threadIdx.x; e < BATCH * 16; e += blockDim.x) {
    int b = e >> 4, j = e & 15;
    int n = n0 + j;                         // < 2800 always
    float ir  = accs[0][b][j] + bih[0 * SDEC + n];
    float iz  = accs[1][b][j] + bih[1 * SDEC + n];
    float inn = accs[2][b][j] + bih[2 * SDEC + n];
    float hr  = accs[3][b][j] + bhh[0 * SDEC + n];
    float hz  = accs[4][b][j] + bhh[1 * SDEC + n];
    float hn  = accs[5][b][j] + bhh[2 * SDEC + n];
    float r = 1.f / (1.f + __expf(-(ir + hr)));
    float z = 1.f / (1.f + __expf(-(iz + hz)));
    float nn = tanhf(inn + r * hn);
    float ho = h_f32[(size_t)b * SDEC_P + n];
    float hv = (1.f - z) * nn + z * ho;
    h_f32_new[(size_t)b * SDEC_P + n] = hv;
    h_bf_new[(size_t)b * SDEC_P + n]  = f2bf(hv);
  }
}

// ---------------- x_next = h @ W_lin_s.T + b ; also init word-decoder state
// block = 256 (8 waves), K=2816 split across waves, LDS reduce
__global__ void sent_linear(const u16* __restrict__ h_bf,     // 32 x 2816
                            const u16* __restrict__ Wlin,     // 320 x 2816
                            const float* __restrict__ blin,   // 300
                            u16* __restrict__ x_s, u16* __restrict__ h_w0,
                            u16* __restrict__ x_w, u64* __restrict__ amax) {
  __shared__ float accs[8][BATCH][16];
  int n0 = blockIdx.x * 16;
  int wave = threadIdx.x >> 5, lane = threadIdx.x & 31;
  const u16* W = Wlin + (size_t)n0 * SDEC_P;
  v8f a0 = {}, a1 = {};
  for (int kc = wave; kc < SDEC_P / 32; kc += 8) {
    int k0 = kc * 32;
    v16bf fb = load_frag(W, SDEC_P, 0, k0, lane);
    v16bf fa = load_frag(h_bf, SDEC_P, 0, k0, lane);
    a0 = WMMA_BF16(fa, fb, a0);
    fa = load_frag(h_bf, SDEC_P, 16, k0, lane);
    a1 = WMMA_BF16(fa, fb, a1);
  }
  int col = lane & 15, rb = (lane & 16) ? 8 : 0;
#pragma unroll
  for (int j = 0; j < 8; ++j) {
    accs[wave][rb + j][col]      = a0[j];
    accs[wave][16 + rb + j][col] = a1[j];
  }
  __syncthreads();
  for (int e = threadIdx.x; e < BATCH * 16; e += blockDim.x) {
    int b = e >> 4, j = e & 15;
    int n = n0 + j;
    float s = 0.f;
#pragma unroll
    for (int w = 0; w < 8; ++w) s += accs[w][b][j];
    s += (n < WDEC) ? blin[n] : 0.f;
    u16 v = f2bf(s);
    x_s[(size_t)b * WDEC_P + n] = v;   // next sentence-step input
    h_w0[(size_t)b * WDEC_P + n] = v;  // initial word-decoder hidden
  }
  if (blockIdx.x == 0) {
    for (int e = threadIdx.x; e < BATCH * EMB_P; e += blockDim.x) {
      int c = e % EMB_P;
      x_w[e] = (c < EMB) ? (u16)0x3F80 : (u16)0;  // x0 = ones
    }
    if (threadIdx.x < BATCH) amax[threadIdx.x] = 0ull;
  }
}

// ---------------- word GRU step ------------------------------------------
__global__ void word_gru_step(const u16* __restrict__ x_bf,   // 32 x 96
                              const u16* __restrict__ h_bf,   // 32 x 320
                              const u16* __restrict__ Wih,    // (3,320,96)
                              const u16* __restrict__ Whh,    // (3,320,320)
                              const float* __restrict__ bih,  // 900
                              const float* __restrict__ bhh,  // 900
                              u16* __restrict__ h_bf_new) {
  __shared__ float accs[6][BATCH][16];
  int n0 = blockIdx.x * 16;                 // 0..304
  int wave = threadIdx.x >> 5, lane = threadIdx.x & 31;
  if (wave < 6) {
    int src = wave / 3, gate = wave % 3;
    const u16* W; const u16* X; int ldk; int nk;
    if (src == 0) { W = Wih + (size_t)(gate * WDEC_P + n0) * EMB_P;  X = x_bf; ldk = EMB_P;  nk = EMB_P / 32; }
    else          { W = Whh + (size_t)(gate * WDEC_P + n0) * WDEC_P; X = h_bf; ldk = WDEC_P; nk = WDEC_P / 32; }
    v8f a0 = {}, a1 = {};
    for (int kc = 0; kc < nk; ++kc) {
      int k0 = kc * 32;
      v16bf fb = load_frag(W, ldk, 0, k0, lane);
      v16bf fa = load_frag(X, ldk, 0, k0, lane);
      a0 = WMMA_BF16(fa, fb, a0);
      fa = load_frag(X, ldk, 16, k0, lane);
      a1 = WMMA_BF16(fa, fb, a1);
    }
    int col = lane & 15, rb = (lane & 16) ? 8 : 0;
#pragma unroll
    for (int j = 0; j < 8; ++j) {
      accs[wave][rb + j][col]      = a0[j];
      accs[wave][16 + rb + j][col] = a1[j];
    }
  }
  __syncthreads();
  for (int e = threadIdx.x; e < BATCH * 16; e += blockDim.x) {
    int b = e >> 4, j = e & 15;
    int n = n0 + j;                         // 0..319; >=300 lanes stay 0
    float br0 = (n < WDEC) ? bih[0 * WDEC + n] : 0.f;
    float br1 = (n < WDEC) ? bih[1 * WDEC + n] : 0.f;
    float br2 = (n < WDEC) ? bih[2 * WDEC + n] : 0.f;
    float bh0 = (n < WDEC) ? bhh[0 * WDEC + n] : 0.f;
    float bh1 = (n < WDEC) ? bhh[1 * WDEC + n] : 0.f;
    float bh2 = (n < WDEC) ? bhh[2 * WDEC + n] : 0.f;
    float ir  = accs[0][b][j] + br0;
    float iz  = accs[1][b][j] + br1;
    float inn = accs[2][b][j] + br2;
    float hr  = accs[3][b][j] + bh0;
    float hz  = accs[4][b][j] + bh1;
    float hn  = accs[5][b][j] + bh2;
    float r = 1.f / (1.f + __expf(-(ir + hr)));
    float z = 1.f / (1.f + __expf(-(iz + hz)));
    float nn = tanhf(inn + r * hn);
    float ho = bf2f(h_bf[(size_t)b * WDEC_P + n]);
    float hv = (1.f - z) * nn + z * ho;
    h_bf_new[(size_t)b * WDEC_P + n] = f2bf(hv);
  }
}

// ---------------- logits + streaming argmax (packed u64 atomicMax) --------
// block = 64 (2 waves), one vocab 16-tile per block
__global__ void word_logits(const u16* __restrict__ h_bf,    // 32 x 320
                            const u16* __restrict__ Wlin,    // 4000 x 320
                            const float* __restrict__ blin,  // 4000
                            float* __restrict__ outp,        // + (s*10+w)*32*4000
                            u64* __restrict__ amax) {
  __shared__ float accs[BATCH][16];
  int n0 = blockIdx.x * 16;
  int wave = threadIdx.x >> 5, lane = threadIdx.x & 31;
  const u16* W = Wlin + (size_t)n0 * WDEC_P;
  v8f acc = {};
  for (int kc = 0; kc < WDEC_P / 32; ++kc) {
    int k0 = kc * 32;
    v16bf fb = load_frag(W, WDEC_P, 0, k0, lane);
    v16bf fa = load_frag(h_bf, WDEC_P, wave * 16, k0, lane);
    acc = WMMA_BF16(fa, fb, acc);
  }
  int col = lane & 15, rb = ((lane & 16) ? 8 : 0) + wave * 16;
#pragma unroll
  for (int j = 0; j < 8; ++j) accs[rb + j][col] = acc[j];
  __syncthreads();
  if (threadIdx.x < BATCH) {
    int b = threadIdx.x;
    float best = -3.402823466e38f;
    int bi = n0;
#pragma unroll
    for (int j = 0; j < 16; ++j) {
      int v = n0 + j;
      float val = accs[b][j] + blin[v];
      outp[(size_t)b * VOCAB + v] = val;
      if (val > best) { best = val; bi = v; }
    }
    u32 bits = __float_as_uint(best);
    u32 hi = (bits & 0x80000000u) ? ~bits : (bits | 0x80000000u);  // order-preserving
    u64 key = ((u64)hi << 32) | (u64)(0xFFFFFFFFu - (u32)bi);      // ties -> first idx
    atomicMax(&amax[b], key);
  }
}

// ---------------- greedy-token embedding gather + argmax reset ------------
__global__ void gather_embed(const u16* __restrict__ embb,  // 4000 x 96
                             u64* __restrict__ amax, u16* __restrict__ x_w) {
  __shared__ int idx[BATCH];
  if (threadIdx.x < BATCH) {
    u64 k = amax[threadIdx.x];
    idx[threadIdx.x] = (int)(0xFFFFFFFFu - (u32)(k & 0xFFFFFFFFull));
  }
  __syncthreads();
  for (int e = threadIdx.x; e < BATCH * EMB_P; e += blockDim.x) {
    int b = e / EMB_P, c = e % EMB_P;
    x_w[e] = embb[(size_t)idx[b] * EMB_P + c];  // padded cols are zero
  }
  __syncthreads();
  if (threadIdx.x < BATCH) amax[threadIdx.x] = 0ull;
}

__global__ void copy_hidden(const float* __restrict__ hf, float* __restrict__ outp) {
  for (int i = blockIdx.x * blockDim.x + threadIdx.x; i < BATCH * SDEC;
       i += gridDim.x * blockDim.x) {
    int b = i / SDEC, n = i % SDEC;
    outp[i] = hf[(size_t)b * SDEC_P + n];
  }
}

extern "C" void kernel_launch(void* const* d_in, const int* in_sizes, int n_in,
                              void* d_out, int out_size, void* d_ws, size_t ws_size,
                              hipStream_t stream) {
  const float* hidden  = (const float*)d_in[0];
  const float* W_ih_s  = (const float*)d_in[1];
  const float* W_hh_s  = (const float*)d_in[2];
  const float* b_ih_s  = (const float*)d_in[3];
  const float* b_hh_s  = (const float*)d_in[4];
  const float* W_lin_s = (const float*)d_in[5];
  const float* b_lin_s = (const float*)d_in[6];
  const float* W_ih_w  = (const float*)d_in[7];
  const float* W_hh_w  = (const float*)d_in[8];
  const float* b_ih_w  = (const float*)d_in[9];
  const float* b_hh_w  = (const float*)d_in[10];
  const float* W_lin_w = (const float*)d_in[11];
  const float* b_lin_w = (const float*)d_in[12];
  const float* emb     = (const float*)d_in[13];
  float* out = (float*)d_out;
  (void)in_sizes; (void)n_in; (void)out_size; (void)ws_size;

  char* p = (char*)d_ws;
  auto carve = [&](size_t bytes) -> void* {
    void* r = (void*)p;
    p += (bytes + 255) & ~(size_t)255;
    return r;
  };
  u16* wihs  = (u16*)carve((size_t)3 * SDEC * WDEC_P * 2);
  u16* whhs  = (u16*)carve((size_t)3 * SDEC * SDEC_P * 2);
  u16* wlins = (u16*)carve((size_t)WDEC_P * SDEC_P * 2);
  u16* wihw  = (u16*)carve((size_t)3 * WDEC_P * EMB_P * 2);
  u16* whhw  = (u16*)carve((size_t)3 * WDEC_P * WDEC_P * 2);
  u16* wlinw = (u16*)carve((size_t)VOCAB * WDEC_P * 2);
  u16* embb  = (u16*)carve((size_t)VOCAB * EMB_P * 2);
  float* hf[2];
  hf[0] = (float*)carve((size_t)BATCH * SDEC_P * 4);
  hf[1] = (float*)carve((size_t)BATCH * SDEC_P * 4);
  u16* hb[2];
  hb[0] = (u16*)carve((size_t)BATCH * SDEC_P * 2);
  hb[1] = (u16*)carve((size_t)BATCH * SDEC_P * 2);
  u16* xs = (u16*)carve((size_t)BATCH * WDEC_P * 2);
  u16* hw[2];
  hw[0] = (u16*)carve((size_t)BATCH * WDEC_P * 2);
  hw[1] = (u16*)carve((size_t)BATCH * WDEC_P * 2);
  u16* xw  = (u16*)carve((size_t)BATCH * EMB_P * 2);
  u64* amax = (u64*)carve((size_t)BATCH * 8);

  // one-time (per call) bf16 weight staging, zero-padded to WMMA tiles
  convert_pad<<<2048, 256, 0, stream>>>(W_ih_s, 3 * SDEC, WDEC, wihs, 3 * SDEC, WDEC_P);
  convert_pad<<<2048, 256, 0, stream>>>(W_hh_s, 3 * SDEC, SDEC, whhs, 3 * SDEC, SDEC_P);
  convert_pad<<<2048, 256, 0, stream>>>(W_lin_s, WDEC, SDEC, wlins, WDEC_P, SDEC_P);
  convert_pad_gates<<<512, 256, 0, stream>>>(W_ih_w, WDEC, EMB, wihw, WDEC_P, EMB_P);
  convert_pad_gates<<<512, 256, 0, stream>>>(W_hh_w, WDEC, WDEC, whhw, WDEC_P, WDEC_P);
  convert_pad<<<2048, 256, 0, stream>>>(W_lin_w, VOCAB, WDEC, wlinw, VOCAB, WDEC_P);
  convert_pad<<<512, 256, 0, stream>>>(emb, VOCAB, EMB, embb, VOCAB, EMB_P);
  init_state<<<256, 256, 0, stream>>>(hidden, hf[0], hb[0], xs);

  const size_t LOGITS = (size_t)MAX_SENT * MAX_WORD * BATCH * VOCAB;
  for (int s = 0; s < MAX_SENT; ++s) {
    int cur = s & 1, nxt = cur ^ 1;
    sent_gru_step<<<SDEC / 16, 256, 0, stream>>>(xs, hb[cur], hf[cur], wihs, whhs,
                                                 b_ih_s, b_hh_s, hf[nxt], hb[nxt]);
    sent_linear<<<WDEC_P / 16, 256, 0, stream>>>(hb[nxt], wlins, b_lin_s, xs, hw[0],
                                                 xw, amax);
    for (int w = 0; w < MAX_WORD; ++w) {
      int wc = w & 1, wn = wc ^ 1;
      word_gru_step<<<WDEC_P / 16, 256, 0, stream>>>(xw, hw[wc], wihw, whhw,
                                                     b_ih_w, b_hh_w, hw[wn]);
      word_logits<<<VOCAB / 16, 64, 0, stream>>>(
          hw[wn], wlinw, b_lin_w,
          out + (size_t)(s * MAX_WORD + w) * BATCH * VOCAB, amax);
      gather_embed<<<1, 128, 0, stream>>>(embb, amax, xw);
    }
  }
  copy_hidden<<<350, 256, 0, stream>>>(hf[0], out + LOGITS);
}
